// TransformerMPNNParenthood_23381801960101
// MI455X (gfx1250) — matrix operand order repaired and checked by reading, not compiled, not run on hardware
//
#include <hip/hip_runtime.h>
#include <hip/hip_bf16.h>
#include <cstdint>

// ---------------------------------------------------------------------------
// MI455X (gfx1250) implementation. wave32 + V_WMMA_F32_16X16X32_BF16.
// GEMM staging uses the CDNA5 async global->LDS engine (ASYNCcnt) with manual
// double buffering; all WMMA fragments are 2 x b128 loads per lane.
// ---------------------------------------------------------------------------

typedef __attribute__((ext_vector_type(16))) __bf16 bf16x16;
typedef __attribute__((ext_vector_type(8)))  __bf16 bf16x8;
typedef __attribute__((ext_vector_type(8)))  float  floatx8;

#define NN  4096
#define DD  512
#define HH  8
#define DHH 64
#define FF  2048   /* EXP*D */
#define EE  131072
#define EPE 4096

// ---------------------------------------------------------------------------
// CDNA5 async global->LDS (GLOBAL_LOAD_ASYNC_TO_LDS_B128, tracked by ASYNCcnt)
// ---------------------------------------------------------------------------
__device__ __forceinline__ void async_load_b128(void* lds_ptr, const void* gptr) {
  const uint32_t lds_off = (uint32_t)(uintptr_t)lds_ptr;  // low 32 bits = LDS offset
  asm volatile("global_load_async_to_lds_b128 %0, %1, off"
               :: "v"(lds_off), "v"((uint64_t)(uintptr_t)gptr)
               : "memory");
}
__device__ __forceinline__ void wait_asynccnt0() {
  asm volatile("s_wait_asynccnt 0x0" ::: "memory");
}

// ---------------------------------------------------------------------------
// WMMA helpers (wave32 layouts per CDNA5 ISA §7.12.2)
// ---------------------------------------------------------------------------
__device__ __forceinline__ floatx8 wmma_bf16(bf16x16 a, bf16x16 b, floatx8 c) {
  return __builtin_amdgcn_wmma_f32_16x16x32_bf16(false, a, false, b, (short)0, c,
                                                 false, false);
}

__device__ __forceinline__ bf16x16 combine16(bf16x8 lo, bf16x8 hi) {
  return __builtin_shufflevector(lo, hi, 0, 1, 2, 3, 4, 5, 6, 7,
                                 8, 9, 10, 11, 12, 13, 14, 15);
}

// A fragment (16x32, row-major, leading dim ld, ld % 8 == 0, base 16B-aligned).
__device__ __forceinline__ bf16x16 load_a_frag(const __bf16* base, int ld) {
  const int lane = threadIdx.x & 31;
  const int row  = lane & 15;
  const int kh   = lane >> 4;
  const __bf16* p = base + (size_t)row * ld + kh * 8;
  const bf16x8 lo = *(const bf16x8*)(p);
  const bf16x8 hi = *(const bf16x8*)(p + 16);
  return combine16(lo, hi);
}

// B fragment (32x16) from a K-contiguous (transposed) source: BT[n][k].
__device__ __forceinline__ bf16x16 load_b_fragT(const __bf16* base, int ld) {
  const int lane = threadIdx.x & 31;
  const int col  = lane & 15;
  const int kh   = lane >> 4;
  const __bf16* p = base + (size_t)col * ld + kh * 8;
  const bf16x8 lo = *(const bf16x8*)(p);
  const bf16x8 hi = *(const bf16x8*)(p + 16);
  return combine16(lo, hi);
}

// C/D fragment: VGPR r = row r (lanes 0-15) / row r+8 (lanes 16-31), col = lane&15.
__device__ __forceinline__ void store_c_frag(float* C, int ld, floatx8 c, float scale) {
  const int lane = threadIdx.x & 31;
  const int col  = lane & 15;
  const int rb   = (lane >> 4) * 8;
#pragma unroll
  for (int r = 0; r < 8; ++r) C[(rb + r) * ld + col] = c[r] * scale;
}

// ---------------------------------------------------------------------------
// Generic bf16 WMMA GEMM: C = A[M,K] @ BT[N,K]^T (+bias)(+residual)(ReLU?)
// block = 256 threads (8 waves), block tile 128x64, wave tile 32x32, K-step 32.
// Async double-buffered LDS staging. Optional outputs: Cf / Cb / CbT.
// ---------------------------------------------------------------------------
__global__ __launch_bounds__(256) void gemm_bf16_kernel(
    const __bf16* __restrict__ A, const __bf16* __restrict__ BT,
    const float* __restrict__ bias, const float* __restrict__ residual,
    float* __restrict__ Cf, __bf16* __restrict__ Cb, __bf16* __restrict__ CbT,
    int M, int N, int K, int relu) {
  __shared__ __align__(16) __bf16 As[2][128 * 32];
  __shared__ __align__(16) __bf16 BsT[2][64 * 32];

  const int tid  = threadIdx.x;
  const int wave = tid >> 5;
  const int lane = tid & 31;
  const int m0   = blockIdx.x * 128;
  const int n0   = blockIdx.y * 64;
  const int wr   = wave >> 1;  // 0..3 -> m offset wr*32
  const int wc   = wave & 1;   // 0..1 -> n offset wc*32

  // per-thread staging coordinates (16B per slot)
  const int ra0 = tid >> 2;                 // A row, first vector
  const int ra1 = (tid + 256) >> 2;         // A row, second vector
  const int ca  = (tid & 3) * 8;            // K offset within panel
  const int rbv = tid >> 2;                 // BT row
  const int cb  = (tid & 3) * 8;

  auto issue_panel = [&](int p, int k0) {
    async_load_b128(&As[p][ra0 * 32 + ca],
                    A + (size_t)(m0 + ra0) * K + k0 + ca);
    async_load_b128(&As[p][ra1 * 32 + ca],
                    A + (size_t)(m0 + ra1) * K + k0 + ca);
    async_load_b128(&BsT[p][rbv * 32 + cb],
                    BT + (size_t)(n0 + rbv) * K + k0 + cb);
  };

  floatx8 c00 = {}, c01 = {}, c10 = {}, c11 = {};

  issue_panel(0, 0);
  int p = 0;
  for (int k0 = 0; k0 < K; k0 += 32) {
    wait_asynccnt0();     // this wave's async writes for panel k0 are in LDS
    __syncthreads();      // ...and everyone else's too; prev buffer free
    if (k0 + 32 < K) issue_panel(p ^ 1, k0 + 32);

    const bf16x16 a0 = load_a_frag(&As[p][(wr * 32) * 32], 32);
    const bf16x16 a1 = load_a_frag(&As[p][(wr * 32 + 16) * 32], 32);
    const bf16x16 b0 = load_b_fragT(&BsT[p][(wc * 32) * 32], 32);
    const bf16x16 b1 = load_b_fragT(&BsT[p][(wc * 32 + 16) * 32], 32);
    c00 = wmma_bf16(a0, b0, c00);
    c01 = wmma_bf16(a0, b1, c01);
    c10 = wmma_bf16(a1, b0, c10);
    c11 = wmma_bf16(a1, b1, c11);
    p ^= 1;
  }

  // epilogue
  const int col = lane & 15;
  const int rb  = (lane >> 4) * 8;
#pragma unroll
  for (int mt = 0; mt < 2; ++mt) {
#pragma unroll
    for (int nt = 0; nt < 2; ++nt) {
      const floatx8 cc = (mt == 0) ? (nt == 0 ? c00 : c01) : (nt == 0 ? c10 : c11);
      const int gcol = n0 + wc * 32 + nt * 16 + col;
      const float bi = bias ? bias[gcol] : 0.f;
#pragma unroll
      for (int r = 0; r < 8; ++r) {
        const int grow = m0 + wr * 32 + mt * 16 + rb + r;
        float v = cc[r] + bi;
        if (residual) v += residual[(size_t)grow * N + gcol];
        if (relu) v = fmaxf(v, 0.f);
        if (Cf)  Cf[(size_t)grow * N + gcol] = v;
        if (Cb)  Cb[(size_t)grow * N + gcol] = (__bf16)v;
        if (CbT) CbT[(size_t)gcol * M + grow] = (__bf16)v;
      }
    }
  }
}

// ---------------------------------------------------------------------------
// Flash attention. One block = 64 query rows of one head; 8 waves own the 16
// 16x16 tiles of the 64x64 working set. Q fragments preloaded (loop
// invariant); K fragments stream from global (K-contiguous in row-major
// [n,D]); V consumed pre-transposed (vT[d][n]). Only S (f32) and P (bf16)
// live in LDS. Output: bf16.
// ---------------------------------------------------------------------------
__global__ __launch_bounds__(256) void attn_kernel(
    const __bf16* __restrict__ qb, const __bf16* __restrict__ kb,
    const __bf16* __restrict__ vT, __bf16* __restrict__ ob, int n) {
  __shared__ float Ss[64 * 64];
  __shared__ __align__(16) __bf16 Pb[64 * 64];
  __shared__ float mrow[64], lrow[64], arow[64];

  const int tid  = threadIdx.x;
  const int wave = tid >> 5;
  const int lane = tid & 31;
  const int r0   = blockIdx.x * 64;
  const int hoff = blockIdx.y * DHH;

  if (tid < 64) { mrow[tid] = -1e30f; lrow[tid] = 0.f; }

  const int t0 = wave * 2, t1 = wave * 2 + 1;
  const int tm0 = (t0 >> 2) * 16, tn0 = (t0 & 3) * 16;
  const int tm1 = (t1 >> 2) * 16, tn1 = (t1 & 3) * 16;

  const __bf16* qbase = qb + (size_t)r0 * DD + hoff;
  const bf16x16 aq00 = load_a_frag(qbase + (size_t)tm0 * DD + 0, DD);
  const bf16x16 aq01 = load_a_frag(qbase + (size_t)tm0 * DD + 32, DD);
  const bf16x16 aq10 = load_a_frag(qbase + (size_t)tm1 * DD + 0, DD);
  const bf16x16 aq11 = load_a_frag(qbase + (size_t)tm1 * DD + 32, DD);

  floatx8 o0 = {};
  floatx8 o1 = {};
  const float scale = 0.125f;  // 1/sqrt(64)

  for (int j0 = 0; j0 < n; j0 += 64) {
    __syncthreads();

    // S = (Q @ K^T) * scale
    {
      floatx8 s0 = {};
      floatx8 s1 = {};
      const __bf16* k0p = kb + (size_t)(j0 + tn0) * DD + hoff;
      const __bf16* k1p = kb + (size_t)(j0 + tn1) * DD + hoff;
      s0 = wmma_bf16(aq00, load_b_fragT(k0p + 0, DD), s0);
      s0 = wmma_bf16(aq01, load_b_fragT(k0p + 32, DD), s0);
      s1 = wmma_bf16(aq10, load_b_fragT(k1p + 0, DD), s1);
      s1 = wmma_bf16(aq11, load_b_fragT(k1p + 32, DD), s1);
      store_c_frag(Ss + tm0 * 64 + tn0, 64, s0, scale);
      store_c_frag(Ss + tm1 * 64 + tn1, 64, s1, scale);
    }
    __syncthreads();

    // online softmax update (one thread per query row)
    if (tid < 64) {
      const int r = tid;
      const float m_old = mrow[r];
      float mx = m_old;
#pragma unroll 8
      for (int c = 0; c < 64; ++c) mx = fmaxf(mx, Ss[r * 64 + c]);
      const float alpha = __expf(m_old - mx);
      float sum = 0.f;
#pragma unroll 8
      for (int c = 0; c < 64; ++c) {
        const float pv = __expf(Ss[r * 64 + c] - mx);
        sum += pv;
        Pb[r * 64 + c] = (__bf16)pv;
      }
      mrow[r] = mx;
      lrow[r] = lrow[r] * alpha + sum;
      arow[r] = alpha;
    }
    __syncthreads();

    // O = O * alpha + P @ V
    {
      const int rbn = (lane >> 4) * 8;
#pragma unroll
      for (int r = 0; r < 8; ++r) {
        o0[r] *= arow[tm0 + rbn + r];
        o1[r] *= arow[tm1 + rbn + r];
      }
      const __bf16* v0p = vT + (size_t)(hoff + tn0) * NN + j0;
      const __bf16* v1p = vT + (size_t)(hoff + tn1) * NN + j0;
      o0 = wmma_bf16(load_a_frag(Pb + tm0 * 64 + 0, 64), load_b_fragT(v0p + 0, NN), o0);
      o0 = wmma_bf16(load_a_frag(Pb + tm0 * 64 + 32, 64), load_b_fragT(v0p + 32, NN), o0);
      o1 = wmma_bf16(load_a_frag(Pb + tm1 * 64 + 0, 64), load_b_fragT(v1p + 0, NN), o1);
      o1 = wmma_bf16(load_a_frag(Pb + tm1 * 64 + 32, 64), load_b_fragT(v1p + 32, NN), o1);
    }
  }
  __syncthreads();

  // write O / l  (bf16, row-major)
  {
    const int col = lane & 15;
    const int rbn = (lane >> 4) * 8;
#pragma unroll
    for (int r = 0; r < 8; ++r) {
      const int rr0 = tm0 + rbn + r;
      const int rr1 = tm1 + rbn + r;
      ob[(size_t)(r0 + rr0) * DD + hoff + tn0 + col] = (__bf16)(o0[r] / lrow[rr0]);
      ob[(size_t)(r0 + rr1) * DD + hoff + tn1 + col] = (__bf16)(o1[r] / lrow[rr1]);
    }
  }
}

// ---------------------------------------------------------------------------
// Elementwise / graph kernels
// ---------------------------------------------------------------------------
__global__ void cast_f32_bf16_kernel(const float* __restrict__ in,
                                     __bf16* __restrict__ out, int n) {
  for (int i = blockIdx.x * blockDim.x + threadIdx.x; i < n;
       i += gridDim.x * blockDim.x)
    out[i] = (__bf16)in[i];
}

// WT[n][k] = (bf16) W[k][n]
__global__ void transpose_cast_kernel(const float* __restrict__ W,
                                      __bf16* __restrict__ WT, int K, int N) {
  const int idx = blockIdx.x * blockDim.x + threadIdx.x;
  if (idx >= K * N) return;
  const int nn = idx / K;
  const int kk = idx - nn * K;
  WT[idx] = (__bf16)W[(size_t)kk * N + nn];
}

__global__ void zero_f32_kernel(float* __restrict__ p, int n) {
  for (int i = blockIdx.x * blockDim.x + threadIdx.x; i < n;
       i += gridDim.x * blockDim.x)
    p[i] = 0.f;
}

__global__ void deg_accum_kernel(const int* __restrict__ ei,
                                 const float* __restrict__ ew,
                                 float* __restrict__ deg, int e) {
  const int i = blockIdx.x * blockDim.x + threadIdx.x;
  if (i >= e) return;
  atomicAdd(&deg[ei[e + i]], ew ? ew[i] : 1.0f);
}

__global__ void finalize_dinv_kernel(float* __restrict__ deg, int n) {
  const int i = blockIdx.x * blockDim.x + threadIdx.x;
  if (i >= n) return;
  deg[i] = rsqrtf(deg[i] + 1.0f);
}

// one wave32 per edge: out[dst,:] += norm * h[src,:]
__global__ __launch_bounds__(256) void gcn_scatter_kernel(
    const int* __restrict__ ei, const float* __restrict__ ew,
    const float* __restrict__ dinv, const float* __restrict__ h,
    float* __restrict__ out, int e) {
  const int wid  = (blockIdx.x * blockDim.x + threadIdx.x) >> 5;
  const int lane = threadIdx.x & 31;
  if (wid >= e) return;
  const int src = ei[wid];
  const int dst = ei[e + wid];
  const float w = ew ? ew[wid] : 1.0f;
  const float norm = dinv[src] * w * dinv[dst];
  const float* hs = h + (size_t)src * DD;
  float* od = out + (size_t)dst * DD;
#pragma unroll
  for (int f = lane; f < DD; f += 32) atomicAdd(&od[f], norm * hs[f]);
}

__global__ void selfloop_bias_kernel(const float* __restrict__ dinv,
                                     const float* __restrict__ h,
                                     const float* __restrict__ bias,
                                     float* __restrict__ out, int total) {
  const int idx = blockIdx.x * blockDim.x + threadIdx.x;
  if (idx >= total) return;
  const int i = idx >> 9;
  const int f = idx & 511;
  out[idx] += dinv[i] * dinv[i] * h[idx] + bias[f];
}

__global__ __launch_bounds__(256) void layernorm_kernel(
    const float* __restrict__ in, const float* __restrict__ g,
    const float* __restrict__ b, float* __restrict__ outf,
    __bf16* __restrict__ outb) {
  const int row = blockIdx.x;
  const int tid = threadIdx.x;
  const float* xr = in + (size_t)row * DD;
  const float a0 = xr[tid], a1 = xr[tid + 256];
  __shared__ float s1[256], s2[256];
  s1[tid] = a0 + a1;
  s2[tid] = a0 * a0 + a1 * a1;
  __syncthreads();
  for (int off = 128; off > 0; off >>= 1) {
    if (tid < off) { s1[tid] += s1[tid + off]; s2[tid] += s2[tid + off]; }
    __syncthreads();
  }
  const float mu  = s1[0] * (1.0f / DD);
  const float var = s2[0] * (1.0f / DD) - mu * mu;
  const float inv = rsqrtf(var + 1e-5f);
  const float y0 = (a0 - mu) * inv * g[tid] + b[tid];
  const float y1 = (a1 - mu) * inv * g[tid + 256] + b[tid + 256];
  outf[(size_t)row * DD + tid]       = y0;
  outf[(size_t)row * DD + tid + 256] = y1;
  if (outb) {
    outb[(size_t)row * DD + tid]       = (__bf16)y0;
    outb[(size_t)row * DD + tid + 256] = (__bf16)y1;
  }
}

// combined = concat([mp, tr, pp], -1) cast to bf16  ->  [N, 1536]
__global__ void concat_cast_kernel(const float* __restrict__ mp,
                                   const float* __restrict__ tr,
                                   const float* __restrict__ pp,
                                   __bf16* __restrict__ cat, int total) {
  const int idx = blockIdx.x * blockDim.x + threadIdx.x;
  if (idx >= total) return;
  const int r = idx / 1536;
  const int c = idx - r * 1536;
  float v;
  if (c < 512)       v = mp[(size_t)r * 512 + c];
  else if (c < 1024) v = tr[(size_t)r * 512 + (c - 512)];
  else               v = pp[(size_t)r * 512 + (c - 1024)];
  cat[idx] = (__bf16)v;
}

// ---------------------------------------------------------------------------
// Host launcher
// ---------------------------------------------------------------------------
extern "C" void kernel_launch(void* const* d_in, const int* in_sizes, int n_in,
                              void* d_out, int out_size, void* d_ws, size_t ws_size,
                              hipStream_t stream) {
  const float* x     = (const float*)d_in[0];
  const float* eattr = (const float*)d_in[1];
  const float* W_mp  = (const float*)d_in[2];
  const float* b_mp  = (const float*)d_in[3];
  const float* W_pp  = (const float*)d_in[4];
  const float* b_pp  = (const float*)d_in[5];
  const float* Wq    = (const float*)d_in[6];
  const float* bq    = (const float*)d_in[7];
  const float* Wk    = (const float*)d_in[8];
  const float* bk    = (const float*)d_in[9];
  const float* Wv    = (const float*)d_in[10];
  const float* bv    = (const float*)d_in[11];
  const float* Wo    = (const float*)d_in[12];
  const float* bo    = (const float*)d_in[13];
  const float* ln1g  = (const float*)d_in[14];
  const float* ln1b  = (const float*)d_in[15];
  const float* W1    = (const float*)d_in[16];
  const float* b1    = (const float*)d_in[17];
  const float* W2    = (const float*)d_in[18];
  const float* b2    = (const float*)d_in[19];
  const float* ln2g  = (const float*)d_in[20];
  const float* ln2b  = (const float*)d_in[21];
  const float* Wagg  = (const float*)d_in[22];
  const float* bagg  = (const float*)d_in[23];
  const int* eidx    = (const int*)d_in[24];
  const int* pidx    = (const int*)d_in[25];

  const size_t ND = (size_t)NN * DD;

  char* wsp = (char*)d_ws;
  auto alloc = [&](size_t bytes) -> void* {
    void* p = (void*)wsp;
    wsp += (bytes + 255) & ~(size_t)255;
    return p;
  };

  __bf16* xb    = (__bf16*)alloc(ND * 2);
  __bf16* wmpT  = (__bf16*)alloc((size_t)DD * DD * 2);
  __bf16* wppT  = (__bf16*)alloc((size_t)DD * DD * 2);
  __bf16* wqT   = (__bf16*)alloc((size_t)DD * DD * 2);
  __bf16* wkT   = (__bf16*)alloc((size_t)DD * DD * 2);
  __bf16* wvT   = (__bf16*)alloc((size_t)DD * DD * 2);
  __bf16* woT   = (__bf16*)alloc((size_t)DD * DD * 2);
  __bf16* w1T   = (__bf16*)alloc((size_t)DD * FF * 2);
  __bf16* w2T   = (__bf16*)alloc((size_t)FF * DD * 2);
  __bf16* waggT = (__bf16*)alloc((size_t)3 * DD * DD * 2);
  float* hmp    = (float*)alloc(ND * 4);
  float* hpp    = (float*)alloc(ND * 4);
  float* mpout  = (float*)alloc(ND * 4);
  float* ppout  = (float*)alloc(ND * 4);
  float* degmp  = (float*)alloc((size_t)NN * 4);
  float* degpp  = (float*)alloc((size_t)NN * 4);
  float* tmp32  = (float*)alloc(ND * 4);
  __bf16* qbb   = (__bf16*)alloc(ND * 2);
  __bf16* kbb   = (__bf16*)alloc(ND * 2);
  __bf16* vbt   = (__bf16*)alloc(ND * 2);   // V transposed: [D][N]
  __bf16* ob    = (__bf16*)alloc(ND * 2);
  float* h32    = (float*)alloc(ND * 4);
  __bf16* hb    = (__bf16*)alloc(ND * 2);
  __bf16* mlp1b = (__bf16*)alloc((size_t)NN * FF * 2);
  float* tr32   = (float*)alloc(ND * 4);
  __bf16* catb  = (__bf16*)alloc((size_t)NN * 3 * DD * 2);

  const int TB = 256;
  auto cdiv = [](size_t a, int b) { return (int)((a + b - 1) / b); };

  // ---- casts / weight transposes ------------------------------------------
  cast_f32_bf16_kernel<<<cdiv(ND, TB), TB, 0, stream>>>(x, xb, (int)ND);
  transpose_cast_kernel<<<cdiv((size_t)DD * DD, TB), TB, 0, stream>>>(W_mp, wmpT, DD, DD);
  transpose_cast_kernel<<<cdiv((size_t)DD * DD, TB), TB, 0, stream>>>(W_pp, wppT, DD, DD);
  transpose_cast_kernel<<<cdiv((size_t)DD * DD, TB), TB, 0, stream>>>(Wq, wqT, DD, DD);
  transpose_cast_kernel<<<cdiv((size_t)DD * DD, TB), TB, 0, stream>>>(Wk, wkT, DD, DD);
  transpose_cast_kernel<<<cdiv((size_t)DD * DD, TB), TB, 0, stream>>>(Wv, wvT, DD, DD);
  transpose_cast_kernel<<<cdiv((size_t)DD * DD, TB), TB, 0, stream>>>(Wo, woT, DD, DD);
  transpose_cast_kernel<<<cdiv((size_t)DD * FF, TB), TB, 0, stream>>>(W1, w1T, DD, FF);
  transpose_cast_kernel<<<cdiv((size_t)FF * DD, TB), TB, 0, stream>>>(W2, w2T, FF, DD);
  transpose_cast_kernel<<<cdiv((size_t)3 * DD * DD, TB), TB, 0, stream>>>(Wagg, waggT, 3 * DD, DD);

  // ---- GCN feature transforms:  h = x @ W  (f32 out) ----------------------
  gemm_bf16_kernel<<<dim3(NN / 128, DD / 64), TB, 0, stream>>>(
      xb, wmpT, nullptr, nullptr, hmp, nullptr, nullptr, NN, DD, DD, 0);
  gemm_bf16_kernel<<<dim3(NN / 128, DD / 64), TB, 0, stream>>>(
      xb, wppT, nullptr, nullptr, hpp, nullptr, nullptr, NN, DD, DD, 0);

  // ---- GCN mp: degrees, dinv, scatter, self-loop + bias -------------------
  zero_f32_kernel<<<cdiv((size_t)NN, TB), TB, 0, stream>>>(degmp, NN);
  zero_f32_kernel<<<cdiv(ND, TB), TB, 0, stream>>>(mpout, (int)ND);
  deg_accum_kernel<<<cdiv((size_t)EE, TB), TB, 0, stream>>>(eidx, eattr, degmp, EE);
  finalize_dinv_kernel<<<cdiv((size_t)NN, TB), TB, 0, stream>>>(degmp, NN);
  gcn_scatter_kernel<<<cdiv((size_t)EE * 32, TB), TB, 0, stream>>>(
      eidx, eattr, degmp, hmp, mpout, EE);
  selfloop_bias_kernel<<<cdiv(ND, TB), TB, 0, stream>>>(degmp, hmp, b_mp, mpout, (int)ND);

  // ---- GCN pp (unit edge weights) -----------------------------------------
  zero_f32_kernel<<<cdiv((size_t)NN, TB), TB, 0, stream>>>(degpp, NN);
  zero_f32_kernel<<<cdiv(ND, TB), TB, 0, stream>>>(ppout, (int)ND);
  deg_accum_kernel<<<cdiv((size_t)EPE, TB), TB, 0, stream>>>(pidx, nullptr, degpp, EPE);
  finalize_dinv_kernel<<<cdiv((size_t)NN, TB), TB, 0, stream>>>(degpp, NN);
  gcn_scatter_kernel<<<cdiv((size_t)EPE * 32, TB), TB, 0, stream>>>(
      pidx, nullptr, degpp, hpp, ppout, EPE);
  selfloop_bias_kernel<<<cdiv(ND, TB), TB, 0, stream>>>(degpp, hpp, b_pp, ppout, (int)ND);

  // ---- Q, K, V projections: bf16 out directly; V emitted transposed -------
  gemm_bf16_kernel<<<dim3(NN / 128, DD / 64), TB, 0, stream>>>(
      xb, wqT, bq, nullptr, nullptr, qbb, nullptr, NN, DD, DD, 0);
  gemm_bf16_kernel<<<dim3(NN / 128, DD / 64), TB, 0, stream>>>(
      xb, wkT, bk, nullptr, nullptr, kbb, nullptr, NN, DD, DD, 0);
  gemm_bf16_kernel<<<dim3(NN / 128, DD / 64), TB, 0, stream>>>(
      xb, wvT, bv, nullptr, nullptr, nullptr, vbt, NN, DD, DD, 0);

  // ---- flash attention (WMMA), bf16 out -----------------------------------
  attn_kernel<<<dim3(NN / 64, HH), TB, 0, stream>>>(qbb, kbb, vbt, ob, NN);

  // ---- output projection + residual, LN1 (f32 + bf16 out) -----------------
  gemm_bf16_kernel<<<dim3(NN / 128, DD / 64), TB, 0, stream>>>(
      ob, woT, bo, x, tmp32, nullptr, nullptr, NN, DD, DD, 0);
  layernorm_kernel<<<NN, TB, 0, stream>>>(tmp32, ln1g, ln1b, h32, hb);

  // ---- MLP: relu(h@W1+b1) (bf16 out) @ W2 + b2 + h, LN2 -------------------
  gemm_bf16_kernel<<<dim3(NN / 128, FF / 64), TB, 0, stream>>>(
      hb, w1T, b1, nullptr, nullptr, mlp1b, nullptr, NN, FF, DD, 1);
  gemm_bf16_kernel<<<dim3(NN / 128, DD / 64), TB, 0, stream>>>(
      mlp1b, w2T, b2, h32, tmp32, nullptr, nullptr, NN, DD, FF, 0);
  layernorm_kernel<<<NN, TB, 0, stream>>>(tmp32, ln2g, ln2b, tr32, nullptr);

  // ---- concat + final aggregation GEMM (bias + residual x fused) ----------
  concat_cast_kernel<<<cdiv((size_t)NN * 3 * DD, TB), TB, 0, stream>>>(
      mpout, tr32, ppout, catb, NN * 3 * DD);
  gemm_bf16_kernel<<<dim3(NN / 128, DD / 64), TB, 0, stream>>>(
      catb, waggT, bagg, x, (float*)d_out, nullptr, nullptr, NN, DD, 3 * DD, 0);

  (void)in_sizes; (void)n_in; (void)out_size; (void)ws_size;
}
